// knn__21002390078197
// MI455X (gfx1250) — matrix operand order repaired and checked by reading, not compiled, not run on hardware
//
#include <hip/hip_runtime.h>
#include <stdint.h>

// ---------------- CDNA5 WMMA types ----------------
typedef __attribute__((ext_vector_type(16))) _Float16 v16h;
typedef __attribute__((ext_vector_type(8)))  float    v8f;
typedef __attribute__((ext_vector_type(2)))  _Float16 h2;

#define NN  30000
#define EE  60000
#define NN2 60000
#define AA  120000
#define EE2 150000
#define BB  1024

static inline int ceil_div_i(int a, int b) { return (a + b - 1) / b; }

// ============================================================================
// Fused NNConv edge kernel:
//   msg[e,:] = sum_{k,i} h[e,k] * x[src[e],i] * w2[k, i*64+o]   (+ b2 folded as k=128)
//   aggr[dst[e],:] += msg[e,:]
// GEMM view: U[16, K] @ Wf[K, 64], K = 129*m_in, A-fragments built in registers.
// 128 threads = 4 waves; wave w owns output columns [16w, 16w+16).
// Steady state per chunk: 1 v_wmma + 8 v_pk_mul_f16 + 2 global_load_b128 + 1 ds_load.
// ============================================================================
template <int MLOG>
__global__ __launch_bounds__(128) void k_edge_nnconv(
    const float* __restrict__ edge_attr,   // [E,10]
    const float* __restrict__ w1,          // [10,128]
    const float* __restrict__ b1,          // [128]
    const _Float16* __restrict__ Wt,       // tiled [nchunks][64][32]
    const _Float16* __restrict__ xb,       // [N, m_in] fp16
    const int* __restrict__ src, const int* __restrict__ dst,
    float* __restrict__ aggr,              // [N,64] fp32 (atomic)
    int E_)
{
  constexpr int M_IN  = 1 << MLOG;
  constexpr int NSUB  = M_IN >> 5;          // 32-wide K sub-chunks per h value
  constexpr int NCH   = 129 * NSUB;         // total K chunks (b2 folded as k=128)

  __shared__ int      s_src[16];
  __shared__ int      s_dst[16];
  __shared__ float    s_ea[160];
  __shared__ float    s_w1[1280];
  __shared__ float    s_b1[128];
  __shared__ _Float16 s_H[16 * 128];
  __shared__ _Float16 s_x[16 * 64];

  const int tid = threadIdx.x;
  const int e0  = blockIdx.x * 16;

  if (tid < 16) {
    int e = e0 + tid;
    s_src[tid] = (e < E_) ? src[e] : 0;
    s_dst[tid] = (e < E_) ? dst[e] : -1;
  }
  for (int t = tid; t < 160; t += 128) {
    long gi = (long)e0 * 10 + t;
    s_ea[t] = (gi < (long)E_ * 10) ? edge_attr[gi] : 0.f;
  }
  for (int t = tid; t < 1280; t += 128) s_w1[t] = w1[t];
  if (tid < 128) s_b1[tid] = b1[tid];
  __syncthreads();

  // h = relu(ea @ w1 + b1) for the 16-edge tile, in LDS as fp16
  for (int t = tid; t < 2048; t += 128) {
    int row = t >> 7, k = t & 127;
    float acc = s_b1[k];
#pragma unroll
    for (int j = 0; j < 10; ++j) acc += s_ea[row * 10 + j] * s_w1[j * 128 + k];
    s_H[t] = (_Float16)fmaxf(acc, 0.f);
  }
  // gather x[src] tile
  for (int t = tid; t < 16 * M_IN; t += 128) {
    int row = t >> MLOG, i = t & (M_IN - 1);
    s_x[t] = xb[(size_t)s_src[row] * M_IN + i];
  }
  __syncthreads();

  const int wv  = tid >> 5;
  const int L   = tid & 31;
  const int n0  = wv * 16;
  const int row = L & 15;
  const int hi  = L >> 4;        // 0 or 1 (lane half)

  // Preload this lane's slice of the x row into registers.
  // A-fragment dword j: K=16*hi+2j (contiguous k order, matches B).
  h2 xr[NSUB * 8];
  {
    const h2* xrow2 = (const h2*)(s_x + row * M_IN);
#pragma unroll
    for (int sub = 0; sub < NSUB; ++sub)
#pragma unroll
      for (int g = 0; g < 2; ++g)
#pragma unroll
        for (int p = 0; p < 4; ++p)
          xr[sub * 8 + g * 4 + p] = xrow2[sub * 16 + hi * 8 + g * 4 + p];
  }

  union BU { v16h v; uint4 q[2]; };
  const _Float16* bbase = Wt + (n0 + row) * 32 + hi * 16;

  v8f acc = {};
  // Pipeline: B double-buffered, h loaded one k ahead.
  BU bcur;
  { const uint4* p0 = (const uint4*)bbase; bcur.q[0] = p0[0]; bcur.q[1] = p0[1]; }
  _Float16 hh = s_H[row * 128];   // h for k=0

  int c = 0;
#pragma unroll 1
  for (int k = 0; k < 129; ++k) {
    _Float16 hnx = (k + 1 < 128) ? s_H[row * 128 + k + 1] : (_Float16)1.f;
    h2 hv; hv[0] = hh; hv[1] = hh;
#pragma unroll
    for (int sub = 0; sub < NSUB; ++sub, ++c) {
      BU bn;
      if (c + 1 < NCH) {
        const uint4* pn = (const uint4*)(bbase + (size_t)(c + 1) * 2048);
        bn.q[0] = pn[0]; bn.q[1] = pn[1];
      } else {
        bn = bcur;
      }
      union { v16h v; h2 p[8]; } a;
#pragma unroll
      for (int j = 0; j < 8; ++j) a.p[j] = xr[sub * 8 + j] * hv;  // v_pk_mul_f16
      acc = __builtin_amdgcn_wmma_f32_16x16x32_f16(false, a.v, false, bcur.v,
                                                   (short)0, acc, false, false);
      bcur = bn;
    }
    hh = hnx;
  }

  // Scatter C fragment: lane L, reg r -> (M = r + 8*hi, n = n0 + L%16)
#pragma unroll
  for (int r = 0; r < 8; ++r) {
    int M = r + hi * 8;
    int d = s_dst[M];
    if (d >= 0) atomicAdd(&aggr[(size_t)d * 64 + n0 + row], acc[r]);
  }
}

// ============================================================================
// Generic 16-row WMMA GEMM:  out = act( A1@W1t [+ A2@W2t] [+ addbuf] [+ bias] )
// A row-major fp16 in global, W tiled [nc][64][32]. Output 64 cols.
// ============================================================================
__global__ __launch_bounds__(128) void k_gemm16(
    const _Float16* __restrict__ A1, int sA1, const _Float16* __restrict__ W1t, int nc1,
    const _Float16* __restrict__ A2, int sA2, const _Float16* __restrict__ W2t, int nc2,
    const float* __restrict__ addbuf, const float* __restrict__ bias, int do_relu,
    float* __restrict__ outf, _Float16* __restrict__ outh, int wpad, int Mtot)
{
  __shared__ _Float16 s_a1[16 * 96];
  __shared__ _Float16 s_a2[16 * 96];

  const int tid = threadIdx.x;
  const int m0  = blockIdx.x * 16;
  const int K1  = nc1 * 32;
  const int K2  = nc2 * 32;

  for (int t = tid; t < 16 * K1; t += 128) {
    int r = t / K1, k = t - r * K1;
    int rowg = m0 + r;
    s_a1[t] = (rowg < Mtot) ? A1[(size_t)rowg * sA1 + k] : (_Float16)0.f;
  }
  if (A2) {
    for (int t = tid; t < 16 * K2; t += 128) {
      int r = t / K2, k = t - r * K2;
      int rowg = m0 + r;
      s_a2[t] = (rowg < Mtot) ? A2[(size_t)rowg * sA2 + k] : (_Float16)0.f;
    }
  }
  __syncthreads();

  const int wv = tid >> 5, L = tid & 31;
  const int n0 = wv * 16, row = L & 15, hi = L >> 4, kb = hi * 8;

  v8f acc = {};
  for (int c = 0; c < nc1; ++c) {
    union { v16h v; uint32_t u[8]; } a;
    const uint32_t* xp = (const uint32_t*)(s_a1 + row * K1 + c * 32);
#pragma unroll
    for (int g = 0; g < 2; ++g)
#pragma unroll
      for (int p = 0; p < 4; ++p) a.u[g * 4 + p] = xp[hi * 8 + g * 4 + p];
    union { v16h v; uint4 q[2]; } b;
    const uint4* bp = (const uint4*)(W1t + (size_t)c * 2048 + (n0 + row) * 32 + hi * 16);
    b.q[0] = bp[0]; b.q[1] = bp[1];
    acc = __builtin_amdgcn_wmma_f32_16x16x32_f16(false, a.v, false, b.v, (short)0, acc, false, false);
  }
  if (A2) {
    for (int c = 0; c < nc2; ++c) {
      union { v16h v; uint32_t u[8]; } a;
      const uint32_t* xp = (const uint32_t*)(s_a2 + row * K2 + c * 32);
#pragma unroll
      for (int g = 0; g < 2; ++g)
#pragma unroll
        for (int p = 0; p < 4; ++p) a.u[g * 4 + p] = xp[hi * 8 + g * 4 + p];
      union { v16h v; uint4 q[2]; } b;
      const uint4* bp = (const uint4*)(W2t + (size_t)c * 2048 + (n0 + row) * 32 + hi * 16);
      b.q[0] = bp[0]; b.q[1] = bp[1];
      acc = __builtin_amdgcn_wmma_f32_16x16x32_f16(false, a.v, false, b.v, (short)0, acc, false, false);
    }
  }

#pragma unroll
  for (int r = 0; r < 8; ++r) {
    int M = r + kb;
    int rowg = m0 + M;
    if (rowg < Mtot) {
      int n = n0 + row;
      float v = acc[r];
      if (addbuf) v += addbuf[(size_t)rowg * 64 + n];
      if (bias)   v += bias[n];
      if (do_relu) v = fmaxf(v, 0.f);
      if (outf) outf[(size_t)rowg * 64 + n] = v;
      if (outh) outh[(size_t)rowg * wpad + n] = (_Float16)v;
    }
  }
  if (outh && wpad > 64) {
    int extra = wpad - 64;
    for (int t = tid; t < 16 * extra; t += 128) {
      int r = t / extra, cpos = 64 + (t - (t / extra) * extra);
      int rowg = m0 + r;
      if (rowg < Mtot) outh[(size_t)rowg * wpad + cpos] = (_Float16)0.f;
    }
  }
}

// ============================================================================
// Weight transforms into the B-fragment tiled layout [c][n(64)][kk(32)]
// ============================================================================
__global__ void k_wf2_transform(const float* __restrict__ w2, const float* __restrict__ b2,
                                _Float16* __restrict__ out, int m_in_log2, int nchunks)
{
  int idx = blockIdx.x * 256 + threadIdx.x;
  int total = nchunks * 2048;
  if (idx >= total) return;
  int c = idx >> 11, rem = idx & 2047, n = rem >> 5, kk = rem & 31;
  int KI = c * 32 + kk;
  int m_in = 1 << m_in_log2;
  int kiw = 128 << m_in_log2;
  float v;
  if (KI < kiw) {
    int k = KI >> m_in_log2, i = KI & (m_in - 1);
    v = w2[(size_t)k * (m_in * 64) + i * 64 + n];
  } else {
    int i = KI - kiw;
    v = (i < m_in) ? b2[i * 64 + n] : 0.f;
  }
  out[idx] = (_Float16)v;
}

__global__ void k_w_transform(const float* __restrict__ w, _Float16* __restrict__ out,
                              int Keff, int nchunks)
{
  int idx = blockIdx.x * 256 + threadIdx.x;
  int total = nchunks * 2048;
  if (idx >= total) return;
  int c = idx >> 11, rem = idx & 2047, n = rem >> 5, kk = rem & 31;
  int K = c * 32 + kk;
  out[idx] = (K < Keff) ? (_Float16)w[(size_t)K * 64 + n] : (_Float16)0.f;
}

// ============================================================================
// Utility / pooling / elementwise kernels
// ============================================================================
__global__ void k_zero(float* p, int n) {
  int i = blockIdx.x * 256 + threadIdx.x;
  if (i < n) p[i] = 0.f;
}
__global__ void k_scale(float* p, float s, int n) {
  int i = blockIdx.x * 256 + threadIdx.x;
  if (i < n) p[i] *= s;
}
__global__ void k_f2h_pad(const float* __restrict__ in, int ws,
                          _Float16* __restrict__ out, int wp, int rows)
{
  int t = blockIdx.x * 256 + threadIdx.x;
  if (t >= rows * wp) return;
  int r = t / wp, c = t - r * wp;
  out[t] = (c < ws) ? (_Float16)in[(size_t)r * ws + c] : (_Float16)0.f;
}
__global__ void k_scatter_add_f(const float* __restrict__ in, const int* __restrict__ idxA,
                                const int* __restrict__ idxB, float* __restrict__ out,
                                int ne, int W, int inS, int outS)
{
  int t = blockIdx.x * 256 + threadIdx.x;
  if (t >= ne * W) return;
  int e = t / W, o = t - e * W;
  int a = idxA ? idxA[e] : e;
  atomicAdd(&out[(size_t)idxB[e] * outS + o], in[(size_t)a * inS + o]);
}
__global__ void k_scatter_add_h(const _Float16* __restrict__ in, const int* __restrict__ idxA,
                                const int* __restrict__ idxB, float* __restrict__ out,
                                int ne, int W, int inS, int outS)
{
  int t = blockIdx.x * 256 + threadIdx.x;
  if (t >= ne * W) return;
  int e = t / W, o = t - e * W;
  int a = idxA ? idxA[e] : e;
  atomicAdd(&out[(size_t)idxB[e] * outS + o], (float)in[(size_t)a * inS + o]);
}
__global__ void k_count(const int* __restrict__ idx, float* __restrict__ cnt, int ne) {
  int t = blockIdx.x * 256 + threadIdx.x;
  if (t < ne) atomicAdd(&cnt[idx[t]], 1.f);
}
// x2 = sums/max(cnt,1) concat iso; fp16 padded to 96 cols
__global__ void k_x2cat(const float* __restrict__ sums, const float* __restrict__ cnt,
                        const float* __restrict__ iso, _Float16* __restrict__ x2h, int n2)
{
  int t = blockIdx.x * 256 + threadIdx.x;
  if (t >= n2 * 96) return;
  int n = t / 96, c = t - n * 96;
  float v;
  if (c < 64)      v = sums[(size_t)n * 64 + c] / fmaxf(cnt[n], 1.f);
  else if (c < 80) v = iso[(size_t)n * 16 + (c - 64)];
  else             v = 0.f;
  x2h[t] = (_Float16)v;
}
// hw = relu(emb_water @ fw_w + fw_b)  [3,64]
__global__ void k_hw(const float* __restrict__ embw, const float* __restrict__ fww,
                     const float* __restrict__ fwb, float* __restrict__ hw)
{
  int t = threadIdx.x;
  if (t >= 192) return;
  int j = t >> 6, o = t & 63;
  float acc = fwb[o];
  for (int k = 0; k < 32; ++k) acc += embw[j * 32 + k] * fww[k * 64 + o];
  hw[j * 64 + o] = fmaxf(acc, 0.f);
}
// inter = tanh(x @ hw^T); x' = inter @ hw ; x_sum += x'; xb = fp16(x')
__global__ __launch_bounds__(64) void k_water(const float* __restrict__ xin,
                                              const float* __restrict__ hw,
                                              float* __restrict__ x_sum,
                                              _Float16* __restrict__ xb)
{
  __shared__ float s_row[64];
  __shared__ float s_hw[192];
  __shared__ float s_d[3];
  int n = blockIdx.x, t = threadIdx.x;
  s_row[t] = xin[(size_t)n * 64 + t];
  for (int i = t; i < 192; i += 64) s_hw[i] = hw[i];
  __syncthreads();
  if (t < 3) {
    float a = 0.f;
    for (int o = 0; o < 64; ++o) a += s_row[o] * s_hw[t * 64 + o];
    s_d[t] = tanhf(a);
  }
  __syncthreads();
  float v = s_d[0] * s_hw[t] + s_d[1] * s_hw[64 + t] + s_d[2] * s_hw[128 + t];
  x_sum[(size_t)n * 64 + t] += v;
  xb[(size_t)n * 64 + t] = (_Float16)v;
}
// final MLP
__global__ void k_mlp1(const float* __restrict__ x1, const float* __restrict__ x2,
                       const float* __restrict__ w, const float* __restrict__ b,
                       float* __restrict__ out)
{
  int t = blockIdx.x * 256 + threadIdx.x;
  if (t >= BB * 64) return;
  int g = t >> 6, o = t & 63;
  float acc = b[o];
  for (int k = 0; k < 64; ++k) acc += x1[g * 64 + k] * w[k * 64 + o];
  for (int k = 0; k < 64; ++k) acc += x2[g * 64 + k] * w[(64 + k) * 64 + o];
  out[t] = fmaxf(acc, 0.f);
}
__global__ void k_mlp2(const float* __restrict__ in, const float* __restrict__ w,
                       const float* __restrict__ b, float* __restrict__ out)
{
  int t = blockIdx.x * 256 + threadIdx.x;
  if (t >= BB * 32) return;
  int g = t >> 5, o = t & 31;
  float acc = b[o];
  for (int k = 0; k < 64; ++k) acc += in[g * 64 + k] * w[k * 32 + o];
  out[t] = fmaxf(acc, 0.f);
}
__global__ void k_mlp3(const float* __restrict__ in, const float* __restrict__ w,
                       const float* __restrict__ b, float* __restrict__ out)
{
  int g = blockIdx.x * 256 + threadIdx.x;
  if (g >= BB) return;
  float acc = b[0];
  for (int k = 0; k < 32; ++k) acc += in[g * 32 + k] * w[k];
  out[g] = acc;
}

// ============================================================================
// Host side: input-order resolution + launch sequence
// ============================================================================
struct InMap {
  int x, ea, iso2, embw;
  int w1[3], b1[3], w2[3], b2[3], root[3], cbias[3], fww[3], fwb[3];
  int c4_rel, c4_root, c4_b, c5_rel, c5_root, c5_b;
  int o1w, o1b, o2w, o2b, o3w, o3b;
  int eidx, batch, asg2, eidx2, batch2;
};

static void fill_sorted_params(InMap& m, int b) {
  m.c4_b = b + 0; m.c4_rel = b + 1; m.c4_root = b + 2;
  m.c5_b = b + 3; m.c5_rel = b + 4; m.c5_root = b + 5;
  for (int l = 0; l < 3; ++l) {
    m.cbias[l] = b + 6 + l;  m.fwb[l] = b + 9 + l;  m.fww[l] = b + 12 + l;
    m.b1[l]    = b + 15 + l; m.b2[l]  = b + 18 + l;
    m.w1[l]    = b + 21 + l; m.w2[l]  = b + 24 + l;
  }
  m.o1b = b + 27; m.o1w = b + 28; m.o2b = b + 29; m.o2w = b + 30;
  m.o3b = b + 31; m.o3w = b + 32;
  for (int l = 0; l < 3; ++l) m.root[l] = b + 33 + l;
}

static void resolve(const int* sz, int n_in, InMap& m) {
  int layout = 0; // 0: insertion order, 1: sorted params, 2: fully sorted
  if (n_in >= 45) {
    if (sz[0] == 2 * AA) layout = 2;
    else if (sz[4] == 1280) layout = 0;
    else layout = 1;
  }
  if (layout == 0) {
    m.x = 0; m.ea = 1; m.iso2 = 2; m.embw = 3;
    for (int l = 0; l < 3; ++l) {
      int b = 4 + 8 * l;
      m.w1[l] = b; m.b1[l] = b + 1; m.w2[l] = b + 2; m.b2[l] = b + 3;
      m.root[l] = b + 4; m.cbias[l] = b + 5; m.fww[l] = b + 6; m.fwb[l] = b + 7;
    }
    m.c4_rel = 28; m.c4_root = 29; m.c4_b = 30;
    m.c5_rel = 31; m.c5_root = 32; m.c5_b = 33;
    m.o1w = 34; m.o1b = 35; m.o2w = 36; m.o2b = 37; m.o3w = 38; m.o3b = 39;
    m.eidx = 40; m.batch = 41; m.asg2 = 42; m.eidx2 = 43; m.batch2 = 44;
  } else if (layout == 1) {
    m.x = 0; m.ea = 1; m.iso2 = 2; m.embw = 3;
    fill_sorted_params(m, 4);
    m.eidx = 40; m.batch = 41; m.asg2 = 42; m.eidx2 = 43; m.batch2 = 44;
  } else {
    m.asg2 = 0; m.batch = 1; m.batch2 = 2; m.ea = 3; m.eidx = 4; m.eidx2 = 5;
    m.embw = 6; m.iso2 = 7;
    fill_sorted_params(m, 8);
    m.x = 44;
  }
}

extern "C" void kernel_launch(void* const* d_in, const int* in_sizes, int n_in,
                              void* d_out, int out_size, void* d_ws, size_t ws_size,
                              hipStream_t stream) {
  InMap im;
  resolve(in_sizes, n_in, im);

  const float* x_in   = (const float*)d_in[im.x];
  const float* ea     = (const float*)d_in[im.ea];
  const float* iso2   = (const float*)d_in[im.iso2];
  const float* embw   = (const float*)d_in[im.embw];
  const int*   src    = (const int*)d_in[im.eidx];
  const int*   dst    = src + EE;
  const int*   batch  = (const int*)d_in[im.batch];
  const int*   arow   = (const int*)d_in[im.asg2];
  const int*   acol   = arow + AA;
  const int*   s2     = (const int*)d_in[im.eidx2];
  const int*   d2     = s2 + EE2;
  const int*   batch2 = (const int*)d_in[im.batch2];
  float* out = (float*)d_out;

  // ---- workspace arena ----
  char* base = (char*)d_ws;
  size_t off = 0;
  auto alloc = [&](size_t bytes) -> char* {
    char* p = base + off;
    off = (off + bytes + 255) & ~(size_t)255;
    return p;
  };
  _Float16* xb0    = (_Float16*)alloc((size_t)NN * 32 * 2);
  _Float16* xb     = (_Float16*)alloc((size_t)NN * 64 * 2);
  _Float16* Wf2    = (_Float16*)alloc((size_t)258 * 2048 * 2);
  _Float16* rootT  = (_Float16*)alloc((size_t)2 * 2048 * 2);
  _Float16* c4relT = (_Float16*)alloc((size_t)3 * 2048 * 2);
  _Float16* c4rooT = (_Float16*)alloc((size_t)3 * 2048 * 2);
  _Float16* c5relT = (_Float16*)alloc((size_t)2 * 2048 * 2);
  _Float16* c5rooT = (_Float16*)alloc((size_t)2 * 2048 * 2);
  float* aggr   = (float*)alloc((size_t)NN * 64 * 4);
  float* xtmp   = (float*)alloc((size_t)NN * 64 * 4);
  float* x_sum  = (float*)alloc((size_t)NN * 64 * 4);
  float* hw     = (float*)alloc(192 * 4);
  float* x_1    = (float*)alloc((size_t)BB * 64 * 4);
  float* cnt    = (float*)alloc((size_t)NN2 * 4);
  float* sums   = (float*)alloc((size_t)NN2 * 64 * 4);      // reused as x2fin
  _Float16* x2h    = (_Float16*)alloc((size_t)NN2 * 96 * 2);
  float*    aggr2  = (float*)alloc((size_t)NN2 * 80 * 4);   // reused as aggr2c
  _Float16* aggr2h = (_Float16*)alloc((size_t)NN2 * 96 * 2);// reused as aggr2ch
  _Float16* x2newh = (_Float16*)alloc((size_t)NN2 * 64 * 2);
  float* x_2 = (float*)alloc((size_t)BB * 64 * 4);
  float* m1  = (float*)alloc((size_t)BB * 64 * 4);
  float* m2  = (float*)alloc((size_t)BB * 32 * 4);
  float*    x2fin   = sums;
  float*    aggr2c  = aggr2;
  _Float16* aggr2ch = aggr2h;
  (void)ws_size; (void)out_size;

  // static graph-conv weight transforms
  k_w_transform<<<ceil_div_i(3 * 2048, 256), 256, 0, stream>>>((const float*)d_in[im.c4_rel], c4relT, 80, 3);
  k_w_transform<<<ceil_div_i(3 * 2048, 256), 256, 0, stream>>>((const float*)d_in[im.c4_root], c4rooT, 80, 3);
  k_w_transform<<<ceil_div_i(2 * 2048, 256), 256, 0, stream>>>((const float*)d_in[im.c5_rel], c5relT, 64, 2);
  k_w_transform<<<ceil_div_i(2 * 2048, 256), 256, 0, stream>>>((const float*)d_in[im.c5_root], c5rooT, 64, 2);

  // initial conversions
  k_f2h_pad<<<ceil_div_i(NN * 32, 256), 256, 0, stream>>>(x_in, 32, xb0, 32, NN);
  k_zero<<<ceil_div_i(NN * 64, 256), 256, 0, stream>>>(x_sum, NN * 64);

  // ---- 3 depths of NNConv + water interaction ----
  for (int l = 0; l < 3; ++l) {
    int mlog = (l == 0) ? 5 : 6;
    int m_in = 1 << mlog;
    int nch  = (129 << mlog) / 32;   // 129 or 258
    const _Float16* xcur = (l == 0) ? xb0 : xb;

    k_wf2_transform<<<ceil_div_i(nch * 2048, 256), 256, 0, stream>>>(
        (const float*)d_in[im.w2[l]], (const float*)d_in[im.b2[l]], Wf2, mlog, nch);
    k_w_transform<<<ceil_div_i((m_in / 32) * 2048, 256), 256, 0, stream>>>(
        (const float*)d_in[im.root[l]], rootT, m_in, m_in / 32);
    k_zero<<<ceil_div_i(NN * 64, 256), 256, 0, stream>>>(aggr, NN * 64);

    if (l == 0) {
      k_edge_nnconv<5><<<ceil_div_i(EE, 16), 128, 0, stream>>>(
          ea, (const float*)d_in[im.w1[l]], (const float*)d_in[im.b1[l]],
          Wf2, xcur, src, dst, aggr, EE);
    } else {
      k_edge_nnconv<6><<<ceil_div_i(EE, 16), 128, 0, stream>>>(
          ea, (const float*)d_in[im.w1[l]], (const float*)d_in[im.b1[l]],
          Wf2, xcur, src, dst, aggr, EE);
    }

    k_gemm16<<<ceil_div_i(NN, 16), 128, 0, stream>>>(
        xcur, m_in, rootT, m_in / 32,
        (const _Float16*)nullptr, 0, (const _Float16*)nullptr, 0,
        aggr, (const float*)d_in[im.cbias[l]], 1,
        xtmp, (_Float16*)nullptr, 0, NN);

    k_hw<<<1, 192, 0, stream>>>(embw, (const float*)d_in[im.fww[l]],
                                (const float*)d_in[im.fwb[l]], hw);
    k_water<<<NN, 64, 0, stream>>>(xtmp, hw, x_sum, xb);
  }

  // xmean = x_sum / 3 (in place)
  k_scale<<<ceil_div_i(NN * 64, 256), 256, 0, stream>>>(x_sum, 1.f / 3.f, NN * 64);

  // graph pooling of node features
  k_zero<<<ceil_div_i(BB * 64, 256), 256, 0, stream>>>(x_1, BB * 64);
  k_scatter_add_f<<<ceil_div_i(NN * 64, 256), 256, 0, stream>>>(
      x_sum, (const int*)nullptr, batch, x_1, NN, 64, 64, 64);

  // avg_pool onto 2-tuples
  k_zero<<<ceil_div_i(NN2 * 64, 256), 256, 0, stream>>>(sums, NN2 * 64);
  k_zero<<<ceil_div_i(NN2, 256), 256, 0, stream>>>(cnt, NN2);
  k_scatter_add_f<<<ceil_div_i(AA * 64, 256), 256, 0, stream>>>(
      x_sum, arow, acol, sums, AA, 64, 64, 64);
  k_count<<<ceil_div_i(AA, 256), 256, 0, stream>>>(acol, cnt, AA);
  k_x2cat<<<ceil_div_i(NN2 * 96, 256), 256, 0, stream>>>(sums, cnt, iso2, x2h, NN2);

  // GraphConv c4: aggr2 = segsum(x2[s2]); x2' = relu(aggr2@rel + x2@root + b)
  k_zero<<<ceil_div_i(NN2 * 80, 256), 256, 0, stream>>>(aggr2, NN2 * 80);
  k_scatter_add_h<<<ceil_div_i(EE2 * 80, 256), 256, 0, stream>>>(
      x2h, s2, d2, aggr2, EE2, 80, 96, 80);
  k_f2h_pad<<<ceil_div_i(NN2 * 96, 256), 256, 0, stream>>>(aggr2, 80, aggr2h, 96, NN2);
  k_gemm16<<<ceil_div_i(NN2, 16), 128, 0, stream>>>(
      aggr2h, 96, c4relT, 3, x2h, 96, c4rooT, 3,
      (const float*)nullptr, (const float*)d_in[im.c4_b], 1,
      (float*)nullptr, x2newh, 64, NN2);

  // GraphConv c5
  k_zero<<<ceil_div_i(NN2 * 64, 256), 256, 0, stream>>>(aggr2c, NN2 * 64);
  k_scatter_add_h<<<ceil_div_i(EE2 * 64, 256), 256, 0, stream>>>(
      x2newh, s2, d2, aggr2c, EE2, 64, 64, 64);
  k_f2h_pad<<<ceil_div_i(NN2 * 64, 256), 256, 0, stream>>>(aggr2c, 64, aggr2ch, 64, NN2);
  k_gemm16<<<ceil_div_i(NN2, 16), 128, 0, stream>>>(
      aggr2ch, 64, c5relT, 2, x2newh, 64, c5rooT, 2,
      (const float*)nullptr, (const float*)d_in[im.c5_b], 1,
      x2fin, (_Float16*)nullptr, 0, NN2);

  // pool 2-tuples to graphs
  k_zero<<<ceil_div_i(BB * 64, 256), 256, 0, stream>>>(x_2, BB * 64);
  k_scatter_add_f<<<ceil_div_i(NN2 * 64, 256), 256, 0, stream>>>(
      x2fin, (const int*)nullptr, batch2, x_2, NN2, 64, 64, 64);

  // output MLP
  k_mlp1<<<ceil_div_i(BB * 64, 256), 256, 0, stream>>>(
      x_1, x_2, (const float*)d_in[im.o1w], (const float*)d_in[im.o1b], m1);
  k_mlp2<<<ceil_div_i(BB * 32, 256), 256, 0, stream>>>(
      m1, (const float*)d_in[im.o2w], (const float*)d_in[im.o2b], m2);
  k_mlp3<<<ceil_div_i(BB, 256), 256, 0, stream>>>(
      m2, (const float*)d_in[im.o3w], (const float*)d_in[im.o3b], out);
}
